// PointNetSAMSGLN_15444702396516
// MI455X (gfx1250) — matrix operand order repaired and checked by reading, not compiled, not run on hardware
//
#include <hip/hip_runtime.h>
#include <hip/hip_bf16.h>
#include <cstdint>

// PointNet++ MSG set-abstraction for MI455X (gfx1250, wave32, WMMA + TDM).
// ~17 GFLOP of 16xC GEMMs -> V_WMMA_F32_16X16X32_F16 with f32 accum.
// Ball-query gather done with TENSOR_LOAD_TO_LDS in gather mode (16 row
// indices per descriptor, LDS padding to hit the 192B A-panel row stride).

#define B_    4
#define N_    16384
#define K_    2048
#define CIN_  64
#define COUT_ 256

typedef __attribute__((ext_vector_type(16))) _Float16 v16h;
typedef __attribute__((ext_vector_type(8)))  _Float16 v8h;
typedef __attribute__((ext_vector_type(8)))  float    v8f;
typedef __attribute__((ext_vector_type(4)))  unsigned int v4u;
typedef __attribute__((ext_vector_type(8)))  int v8i;
typedef __attribute__((ext_vector_type(4)))  int v4i;

#if defined(__HIP_DEVICE_COMPILE__) && defined(__has_builtin)
#if __has_builtin(__builtin_amdgcn_tensor_load_to_lds) && \
    __has_builtin(__builtin_amdgcn_s_wait_tensorcnt)
#define HAVE_TDM 1
#endif
#endif
#ifndef HAVE_TDM
#define HAVE_TDM 0
#endif

// ---------------- WMMA fragment helpers (layouts per CDNA5 ISA) ----

__device__ inline v16h make_v16(v8h lo, v8h hi) {
  v16h r;
#pragma unroll
  for (int i = 0; i < 8; ++i) { r[i] = lo[i]; r[i + 8] = hi[i]; }
  return r;
}

// A (16xK f16, row-major source, row stride `stride` halves):
// lane L holds row L&15; kh=L>>4 selects halves {kk+kh*8..+7} and {kk+16+kh*8..+7}.
__device__ inline v16h load_a_frag(const _Float16* base, int stride, int lane, int kk) {
  const int row = lane & 15, kh = lane >> 4;
  const _Float16* p = base + (size_t)row * stride + kk + kh * 8;
  v8h lo = *(const v8h*)(p);
  v8h hi = *(const v8h*)(p + 16);
  return make_v16(lo, hi);
}

// B (32x16 f16) for X*W^T: B[k][n] = W[n][k]. Lane holds column n=lane&15
// (= weight row nt*16+n), kh=lane>>4 selects halves {kk+kh*8..+7} and {kk+16+kh*8..+7}.
__device__ inline v16h load_b_frag(const _Float16* w, int wstride, int lane, int nt, int kk) {
  const int n = lane & 15, kb = (lane >> 4) * 8;
  const _Float16* p = w + (size_t)(nt * 16 + n) * wstride + kk + kb;
  v8h lo = *(const v8h*)(p);
  v8h hi = *(const v8h*)(p + 16);
  return make_v16(lo, hi);
}

// C/D (16x16 f32): VGPR i -> row i + (lane>>4)*8, col = lane&15.
__device__ inline void store_c_frag(float* c, int stride, int lane, int nt, v8f acc) {
  const int col = nt * 16 + (lane & 15);
  const int rb = (lane >> 4) * 8;
#pragma unroll
  for (int i = 0; i < 8; ++i) c[(size_t)(rb + i) * stride + col] = acc[i];
}

__device__ inline v8f wmma_step(v16h a, v16h b, v8f c) {
  return __builtin_amdgcn_wmma_f32_16x16x32_f16(false, a, false, b, (short)0, c,
                                                false, false);
}

// One MLP layer for a 16-row tile: C[16xNTOT] = A[16xKTOT] * W^T
template <int KTOT, int NTOT>
__device__ inline void mlp_layer(const _Float16* a_base, int astride,
                                 const _Float16* w, float* cbuf, int cstride,
                                 int lane) {
#pragma unroll
  for (int nt = 0; nt < NTOT / 16; ++nt) {
    v8f acc = {};
#pragma unroll
    for (int kk = 0; kk < KTOT; kk += 32) {
      v16h a = load_a_frag(a_base, astride, lane, kk);
      v16h b = load_b_frag(w, KTOT, lane, nt, kk);
      acc = wmma_step(a, b, acc);
    }
    store_c_frag(cbuf, cstride, lane, nt, acc);
  }
}

// ---------------- LayerNorm helpers (16 rows, 2 lanes per row) ----------------

__device__ inline float gelu_exact(float x) {
  return 0.5f * x * (1.0f + erff(x * 0.70710678118654752f));
}

__device__ inline void ln_rows_f16(const float* cbuf, int cstride, int C,
                                   const float* g, const float* bt,
                                   _Float16* obuf, int ostride, int lane,
                                   bool do_gelu) {
  const int row = lane & 15, half = lane >> 4;
  const int per = C >> 1;
  const float* r = cbuf + (size_t)row * cstride + half * per;
  float s = 0.f, s2 = 0.f;
  for (int i = 0; i < per; ++i) { float x = r[i]; s += x; s2 += x * x; }
  s  += __shfl_xor(s, 16);
  s2 += __shfl_xor(s2, 16);
  const float m = s / (float)C;
  const float rstd = rsqrtf(s2 / (float)C - m * m + 1e-6f);
  _Float16* o = obuf + (size_t)row * ostride + half * per;
  const float* gg = g + half * per;
  const float* bb = bt + half * per;
  for (int i = 0; i < per; ++i) {
    float x = (r[i] - m) * rstd * gg[i] + bb[i];
    if (do_gelu) x = gelu_exact(x);
    o[i] = (_Float16)x;
  }
}

__device__ inline void ln_rows_f32(float* cbuf, int cstride, int C,
                                   const float* g, const float* bt, int lane) {
  const int row = lane & 15, half = lane >> 4;
  const int per = C >> 1;
  float* r = cbuf + (size_t)row * cstride + half * per;
  float s = 0.f, s2 = 0.f;
  for (int i = 0; i < per; ++i) { float x = r[i]; s += x; s2 += x * x; }
  s  += __shfl_xor(s, 16);
  s2 += __shfl_xor(s2, 16);
  const float m = s / (float)C;
  const float rstd = rsqrtf(s2 / (float)C - m * m + 1e-6f);
  const float* gg = g + half * per;
  const float* bb = bt + half * per;
  for (int i = 0; i < per; ++i) r[i] = (r[i] - m) * rstd * gg[i] + bb[i];
}

// ---------------- TDM gather: 16 feature rows -> LDS A-panel ------------------
// One TENSOR_LOAD_TO_LDS per 16-sample tile. Gather mode, 16-bit indices,
// tile_dim0 = 64 f16 (128B row), LDS padding 16 DWORDs after every 32 DWORDs
// stored -> destination row stride 192B (== sA row stride of 96 halves).

#if HAVE_TDM
__device__ inline void tdm_gather16(const _Float16* tensor_base,
                                    unsigned lds_addr, int myidx, int lane) {
  int dws[8];
#pragma unroll
  for (int j = 0; j < 8; ++j) {
    const int lo = __shfl(myidx, 2 * j);
    const int hi = __shfl(myidx, 2 * j + 1);
    dws[j] = __builtin_amdgcn_readfirstlane((lo & 0xFFFF) | (hi << 16));
  }
  const unsigned long long ga = (unsigned long long)(size_t)tensor_base;
  v4u g0;
  g0[0] = 0x80000001u;                       // gather_mode=1, 16b idx, count=1
  g0[1] = lds_addr;                          // LDS dest (bytes)
  g0[2] = (unsigned)ga;                      // global_addr[31:0]
  g0[3] = (unsigned)(ga >> 32) | (2u << 30); // global_addr[56:32] | type=2
  v8i g1;
  g1[0] = (int)0x1F110000;   // data_size=2B, pad_en, interval=32dw, amount=16dw
  g1[1] = 64 << 16;          // tensor_dim0 = 64 elements per row
  g1[2] = (int)(16384u << 16); // tensor_dim1 = N (row-index bound)
  g1[3] = 64 << 16;          // tile_dim0 = 64
  g1[4] = 16;                // tile_dim1 = 16 valid indices
  g1[5] = 64;                // tensor_dim0_stride = 64
  g1[6] = 0;
  g1[7] = 0;
  v4i g2; v4i g3;
#pragma unroll
  for (int j = 0; j < 4; ++j) { g2[j] = dws[j]; g3[j] = dws[4 + j]; }
#if __clang_major__ >= 23
  v8i zero = {};
  __builtin_amdgcn_tensor_load_to_lds(g0, g1, g2, g3, zero, 0);
#else
  __builtin_amdgcn_tensor_load_to_lds(g0, g1, g2, g3, 0);
#endif
}
#endif

// ---------------- Pre-pass kernels -------------------------------------------

// features (B,64,N) f32 -> featT (B,N,64) f16 (contiguous 128B row per point)
__global__ void __launch_bounds__(256)
feat_to_f16_kernel(const float* __restrict__ f, _Float16* __restrict__ ft) {
  const int t = blockIdx.x * 256 + threadIdx.x;  // (b,n) pair
  const int b = t >> 14, n = t & (N_ - 1);
  const float* src = f + (size_t)b * CIN_ * N_ + n;
  _Float16* dst = ft + (size_t)t * CIN_;
#pragma unroll
  for (int cc = 0; cc < 8; ++cc) {
    v8h v;
#pragma unroll
    for (int j = 0; j < 8; ++j) v[j] = (_Float16)src[(size_t)(cc * 8 + j) * N_];
    *(v8h*)(dst + cc * 8) = v;
  }
}

// w1 (64,67) f32 -> (64,96) f16 with column permute: [feat(64)|gp(3)|0-pad]
__global__ void __launch_bounds__(256)
conv_w1_kernel(const float* __restrict__ w, _Float16* __restrict__ wh) {
  const int t = blockIdx.x * 256 + threadIdx.x;
  if (t >= 64 * 96) return;
  const int r = t / 96, c = t % 96;
  float v = 0.f;
  if (c < 64)      v = w[r * 67 + 3 + c];    // feature columns
  else if (c < 67) v = w[r * 67 + (c - 64)]; // gp columns
  wh[t] = (_Float16)v;
}

__global__ void __launch_bounds__(256)
conv_w_kernel(const float* __restrict__ w, _Float16* __restrict__ wh, int total) {
  const int t = blockIdx.x * 256 + threadIdx.x;
  if (t < total) wh[t] = (_Float16)w[t];
}

// ---------------- Ball query: 1 wave32 per centroid, both radii in one pass --

__global__ void __launch_bounds__(128)
ball_query_kernel(const float* __restrict__ points,
                  const float* __restrict__ cents,
                  int* __restrict__ idx1, int* __restrict__ idx2) {
  const int wave = threadIdx.x >> 5, lane = threadIdx.x & 31;
  const int cid = blockIdx.x * 4 + wave;          // [0, B*K)
  const int b = cid >> 11;                        // K_ = 2048
  const float cx = cents[cid * 3 + 0];
  const float cy = cents[cid * 3 + 1];
  const float cz = cents[cid * 3 + 2];
  const float* pb = points + (size_t)b * N_ * 3;
  int* o1 = idx1 + (size_t)cid * 16;
  int* o2 = idx2 + (size_t)cid * 32;
  const float r1sq = 0.4f * 0.4f, r2sq = 0.8f * 0.8f;
  int cnt1 = 0, cnt2 = 0, first1 = -1, first2 = -1;
  const unsigned lt = (1u << lane) - 1u;
  for (int base = 0; base < N_; base += 32) {
    if (cnt1 >= 16 && cnt2 >= 32) break;          // typical exit after ~200 pts
    const int pre = (base + 256 < N_) ? base + 256 : base;
    __builtin_prefetch(pb + (size_t)pre * 3, 0, 0);  // global_prefetch_b8
    const float* pp = pb + (size_t)(base + lane) * 3;
    const float dx = pp[0] - cx, dy = pp[1] - cy, dz = pp[2] - cz;
    const float d2 = dx * dx + dy * dy + dz * dz;
    const unsigned m1 = (unsigned)__ballot(d2 < r1sq);
    const unsigned m2 = (unsigned)__ballot(d2 < r2sq);
    if (first1 < 0 && m1) first1 = base + __ffs(m1) - 1;
    if (first2 < 0 && m2) first2 = base + __ffs(m2) - 1;
    if (d2 < r1sq) {
      const int pos = cnt1 + __popc(m1 & lt);
      if (pos < 16) o1[pos] = base + lane;
    }
    if (d2 < r2sq) {
      const int pos = cnt2 + __popc(m2 & lt);
      if (pos < 32) o2[pos] = base + lane;
    }
    cnt1 += __popc(m1);
    cnt2 += __popc(m2);
  }
  // pad with first hit; if no hit at all, reference forces idx==0
  const int pad1 = first1 >= 0 ? first1 : 0;
  const int pad2 = first2 >= 0 ? first2 : 0;
  if (lane < 16 && lane >= cnt1) o1[lane] = pad1;
  if (lane >= cnt2) o2[lane] = pad2;
}

// ---------------- Fused gather + 3-layer MLP + LN + maxpool ------------------
// 1 wave per centroid, 4 waves / block. Per 16-sample M-tile:
//   TDM-gather->sA(f16) -> WMMA -> sC(f32) -> LN -> sA(f16) -> ... -> maxpool.

template <int NS, int C2>
__global__ void __launch_bounds__(128)
branch_mlp_kernel(const _Float16* __restrict__ featT,
                  const float* __restrict__ points,
                  const float* __restrict__ cents,
                  const int* __restrict__ idxbuf,
                  const _Float16* __restrict__ w1h, const float* __restrict__ g1,
                  const float* __restrict__ b1,
                  const _Float16* __restrict__ w2h, const float* __restrict__ g2,
                  const float* __restrict__ b2,
                  const _Float16* __restrict__ w3h, const float* __restrict__ g3,
                  const float* __restrict__ b3,
                  _Float16* __restrict__ ybuf, float inv_r, int ch_base) {
  constexpr int C1 = 64, C3 = 128, K1 = 96, K2 = 64, K3 = C2;
  __shared__ _Float16 sA[4][16][96];   // 12 KB  (A panel, K padded to 96)
  __shared__ float    sC[4][16][128];  // 32 KB  (f32 C staging)
  const int wave = threadIdx.x >> 5, lane = threadIdx.x & 31;
  const int cid = blockIdx.x * 4 + wave;
  const int b = cid >> 11;
  const float cx = cents[cid * 3 + 0];
  const float cy = cents[cid * 3 + 1];
  const float cz = cents[cid * 3 + 2];
  const int* idx = idxbuf + (size_t)cid * NS;

  float vmax[4];
#pragma unroll
  for (int j = 0; j < 4; ++j) vmax[j] = -3.0e38f;

  _Float16* A = &sA[wave][0][0];
  float* C = &sC[wave][0][0];

#pragma unroll
  for (int mt = 0; mt < NS / 16; ++mt) {
    const int row = lane & 15, halfsel = lane >> 4;
    const int pi = idx[mt * 16 + row];
#if HAVE_TDM
    // One tensor-DMA gather per tile: 16 rows of 64xf16 into sA (192B stride).
    tdm_gather16(featT + (size_t)b * N_ * 64, (unsigned)(size_t)A, pi, lane);
#else
    {  // fallback: 2 lanes per row copy the 128B f16 feature row
      const _Float16* src = featT + ((size_t)b * N_ + pi) * 64;
      const v8h* sp = (const v8h*)(src + halfsel * 32);
      v8h t0 = sp[0], t1 = sp[1], t2 = sp[2], t3 = sp[3];
      v8h* dp = (v8h*)(A + row * 96 + halfsel * 32);
      dp[0] = t0; dp[1] = t1; dp[2] = t2; dp[3] = t3;
    }
#endif
    if (halfsel == 0) {  // gp columns + zero K-padding (TDM skips these)
      const float* pp = points + ((size_t)b * N_ + pi) * 3;
      A[row * 96 + 64] = (_Float16)((pp[0] - cx) * inv_r);
      A[row * 96 + 65] = (_Float16)((pp[1] - cy) * inv_r);
      A[row * 96 + 66] = (_Float16)((pp[2] - cz) * inv_r);
#pragma unroll
      for (int c = 67; c < 96; ++c) A[row * 96 + c] = (_Float16)0.f;
    }
#if HAVE_TDM
    __builtin_amdgcn_s_wait_tensorcnt(0);
#endif
    __syncthreads();
    mlp_layer<K1, C1>(A, 96, w1h, C, 128, lane);       // 67(->96) -> 64
    __syncthreads();
    ln_rows_f16(C, 128, C1, g1, b1, A, 96, lane, false);
    __syncthreads();
    mlp_layer<K2, C2>(A, 96, w2h, C, 128, lane);       // 64 -> C2
    __syncthreads();
    ln_rows_f16(C, 128, C2, g2, b2, A, 96, lane, true);  // LN + GELU
    __syncthreads();
    mlp_layer<K3, C3>(A, 96, w3h, C, 128, lane);       // C2 -> 128
    __syncthreads();
    ln_rows_f32(C, 128, C3, g3, b3, lane);
    __syncthreads();
#pragma unroll
    for (int j = 0; j < 4; ++j) {  // maxpool over 16 sample rows
      const int col = lane * 4 + j;
      float mx = vmax[j];
#pragma unroll
      for (int r = 0; r < 16; ++r) mx = fmaxf(mx, C[r * 128 + col]);
      vmax[j] = mx;
    }
    __syncthreads();
  }
  _Float16* yo = ybuf + (size_t)cid * COUT_ + ch_base;
#pragma unroll
  for (int j = 0; j < 4; ++j) yo[lane * 4 + j] = (_Float16)vmax[j];
}

// ---------------- Aggregation: 256->256 GEMM + LN + GELU + transpose ---------
// 1 wave per 16 centroids; A rows straight from L2-resident ybuf (f16).

__global__ void __launch_bounds__(64)
agg_kernel(const _Float16* __restrict__ ybuf, const _Float16* __restrict__ wah,
           const float* __restrict__ ga, const float* __restrict__ ba,
           float* __restrict__ out) {
  __shared__ float sC[2][16][256];  // 32 KB
  const int wave = threadIdx.x >> 5, lane = threadIdx.x & 31;
  const int tile = blockIdx.x * 2 + wave;
  const int cid0 = tile * 16;
  const int b = cid0 >> 11;
  const int k0 = cid0 & (K_ - 1);
  const _Float16* A = ybuf + (size_t)cid0 * COUT_;
  float* C = &sC[wave][0][0];
#pragma unroll
  for (int nt = 0; nt < 16; ++nt) {
    v8f acc = {};
#pragma unroll
    for (int kk = 0; kk < 256; kk += 32) {
      v16h a = load_a_frag(A, 256, lane, kk);
      v16h bm = load_b_frag(wah, 256, lane, nt, kk);
      acc = wmma_step(a, bm, acc);
    }
    store_c_frag(C, 256, lane, nt, acc);
  }
  __syncthreads();
  const int row = lane & 15, half = lane >> 4;
  const float* r = C + row * 256 + half * 128;
  float s = 0.f, s2 = 0.f;
  for (int i = 0; i < 128; ++i) { float x = r[i]; s += x; s2 += x * x; }
  s  += __shfl_xor(s, 16);
  s2 += __shfl_xor(s2, 16);
  const float m = s * (1.0f / 256.0f);
  const float rstd = rsqrtf(s2 * (1.0f / 256.0f) - m * m + 1e-6f);
  const int k = k0 + row;
  for (int i = 0; i < 128; ++i) {
    const int c = half * 128 + i;
    float x = (r[i] - m) * rstd * ga[c] + ba[c];
    x = gelu_exact(x);
    out[((size_t)b * COUT_ + c) * K_ + k] = x;  // (B, 256, K)
  }
}

// ---------------- Host launcher ----------------------------------------------
// Input flattening (setup_inputs dict order):
//  0 points  1 features  2 centroids
//  branch0: 3 w1, 4 g1, 5 b1, 6 w2, 7 g2, 8 b2,  9 w3, 10 g3, 11 b3
//  branch1: 12 w1,13 g1,14 b1,15 w2,16 g2,17 b2, 18 w3, 19 g3, 20 b3
//  agg:     21 w, 22 g, 23 b

extern "C" void kernel_launch(void* const* d_in, const int* in_sizes, int n_in,
                              void* d_out, int out_size, void* d_ws,
                              size_t ws_size, hipStream_t stream) {
  (void)in_sizes; (void)n_in; (void)out_size; (void)ws_size;
  const float* points = (const float*)d_in[0];
  const float* features = (const float*)d_in[1];
  const float* cents = (const float*)d_in[2];

  char* ws = (char*)d_ws;
  size_t off = 0;
  auto alloc = [&](size_t bytes) -> void* {
    void* p = ws + off;
    off = (off + bytes + 255) & ~(size_t)255;
    return p;
  };
  _Float16* featT = (_Float16*)alloc((size_t)B_ * N_ * CIN_ * 2);  // 8 MB
  int* idx1 = (int*)alloc((size_t)B_ * K_ * 16 * 4);
  int* idx2 = (int*)alloc((size_t)B_ * K_ * 32 * 4);
  _Float16* w1h0 = (_Float16*)alloc(64 * 96 * 2);
  _Float16* w1h1 = (_Float16*)alloc(64 * 96 * 2);
  _Float16* w2h0 = (_Float16*)alloc(64 * 64 * 2);
  _Float16* w2h1 = (_Float16*)alloc(96 * 64 * 2);
  _Float16* w3h0 = (_Float16*)alloc(128 * 64 * 2);
  _Float16* w3h1 = (_Float16*)alloc(128 * 96 * 2);
  _Float16* wah  = (_Float16*)alloc(256 * 256 * 2);
  _Float16* ybuf = (_Float16*)alloc((size_t)B_ * K_ * COUT_ * 2);  // 4 MB

  feat_to_f16_kernel<<<B_ * N_ / 256, 256, 0, stream>>>(features, featT);
  conv_w1_kernel<<<(64 * 96 + 255) / 256, 256, 0, stream>>>(
      (const float*)d_in[3], w1h0);
  conv_w1_kernel<<<(64 * 96 + 255) / 256, 256, 0, stream>>>(
      (const float*)d_in[12], w1h1);
  conv_w_kernel<<<(64 * 64 + 255) / 256, 256, 0, stream>>>(
      (const float*)d_in[6], w2h0, 64 * 64);
  conv_w_kernel<<<(96 * 64 + 255) / 256, 256, 0, stream>>>(
      (const float*)d_in[15], w2h1, 96 * 64);
  conv_w_kernel<<<(128 * 64 + 255) / 256, 256, 0, stream>>>(
      (const float*)d_in[9], w3h0, 128 * 64);
  conv_w_kernel<<<(128 * 96 + 255) / 256, 256, 0, stream>>>(
      (const float*)d_in[18], w3h1, 128 * 96);
  conv_w_kernel<<<(256 * 256 + 255) / 256, 256, 0, stream>>>(
      (const float*)d_in[21], wah, 256 * 256);

  ball_query_kernel<<<B_ * K_ / 4, 128, 0, stream>>>(points, cents, idx1, idx2);

  branch_mlp_kernel<16, 64><<<B_ * K_ / 4, 128, 0, stream>>>(
      featT, points, cents, idx1, w1h0, (const float*)d_in[4],
      (const float*)d_in[5], w2h0, (const float*)d_in[7],
      (const float*)d_in[8], w3h0, (const float*)d_in[10],
      (const float*)d_in[11], ybuf, 1.0f / 0.4f, 0);
  branch_mlp_kernel<32, 96><<<B_ * K_ / 4, 128, 0, stream>>>(
      featT, points, cents, idx2, w1h1, (const float*)d_in[13],
      (const float*)d_in[14], w2h1, (const float*)d_in[16],
      (const float*)d_in[17], w3h1, (const float*)d_in[19],
      (const float*)d_in[20], ybuf, 1.0f / 0.8f, 128);

  agg_kernel<<<(B_ * K_ / 16) / 2, 64, 0, stream>>>(
      ybuf, wah, (const float*)d_in[22], (const float*)d_in[23],
      (float*)d_out);
}
